// TemporalCrossTransformer_59700045414915
// MI455X (gfx1250) — matrix operand order repaired and checked by reading, not compiled, not run on hardware
//
#include <hip/hip_runtime.h>

#define IN_DIM   2048
#define OUT_DIM  1152
#define SEQ_LEN  8
#define WAY      5
#define SHOT     5
#define NQ       100
#define S_ROWS     (WAY * SHOT * SEQ_LEN)   // 200
#define S_ROWS_PAD 224                      // 7 tiles of 32
#define Q_ROWS     (NQ * SEQ_LEN)           // 800 = 25 tiles of 32
#define KM         (SHOT * SEQ_LEN)         // 40 support frames per class

typedef __attribute__((ext_vector_type(16))) __bf16 v16bf;
typedef __attribute__((ext_vector_type(8)))  float  v8f;

union Frag16 {
  uint4 q[2];
  v16bf v;
};
static_assert(sizeof(Frag16) == 32, "frag size");

__device__ __forceinline__ unsigned short f2bf(float f) {
  union { float f; unsigned int u; } x; x.f = f;
  unsigned int u = x.u;
  u += 0x7FFFu + ((u >> 16) & 1u);   // round-to-nearest-even
  return (unsigned short)(u >> 16);
}

// WGP-scope prefetch (default scope 0): pulls the line into ALL cache levels,
// including WGP$/L0 -- unlike the builtin's SE/SYS-scope lowering which stops
// at GL2 (where this data already resides). No VGPR return, no counters.
__device__ __forceinline__ void prefetch_wgp(const void* p) {
  asm volatile("global_prefetch_b8 %0, off" :: "v"(p) : "memory");
}

// ---------------------------------------------------------------------------
// Prep: activations + positional encoding -> bf16 (rows padded with zeros)
// ---------------------------------------------------------------------------
__global__ void prep_x_kernel(const float* __restrict__ src, const float* __restrict__ pe,
                              unsigned short* __restrict__ dst, int rows, int padRows) {
  int idx = blockIdx.x * blockDim.x + threadIdx.x;
  int total = padRows * IN_DIM;
  if (idx >= total) return;
  int row = idx / IN_DIM;
  int k   = idx - row * IN_DIM;
  float v = 0.0f;
  if (row < rows) v = src[idx] + pe[(row & (SEQ_LEN - 1)) * IN_DIM + k];
  dst[idx] = f2bf(v);
}

__global__ void cvt_bf16_kernel(const float* __restrict__ src,
                                unsigned short* __restrict__ dst, int n) {
  int idx = blockIdx.x * blockDim.x + threadIdx.x;
  if (idx < n) dst[idx] = f2bf(src[idx]);
}

// ---------------------------------------------------------------------------
// WMMA GEMM: Y[M, OUT_DIM] = X[Mpad, IN_DIM] * W[OUT_DIM, IN_DIM]^T + bias
// block = 128 threads = 4 waves. Each wave computes a 32x32 output block
// (2 M-tiles x 2 N-tiles, 4 accumulators) -> A and B fragments each reused
// twice: 8 b128 loads per 4 v_wmma (2 loads/WMMA). Outer K-step of 64 issues
// one WGP-scope prefetch per B row per body (matches cacheline consumption).
// grid = (OUT_DIM/128 = 9, Mpad/32)
// ---------------------------------------------------------------------------
__global__ __launch_bounds__(128)
void gemm_bf16_kernel(const unsigned short* __restrict__ X,
                      const unsigned short* __restrict__ W,
                      const float* __restrict__ bias,
                      float* __restrict__ Y, int M) {
  const int lane    = threadIdx.x & 31;
  const int wave    = threadIdx.x >> 5;
  const int rowBase = blockIdx.y * 32;
  const int colBase = blockIdx.x * 128 + wave * 32;

  // A fragment (16x32 bf16): lane m holds K [ba..ba+7] and [ba+16..ba+23], ba=(lane>=16)?8:0
  const unsigned short* arow0 = X + (size_t)(rowBase + (lane & 15)) * IN_DIM + ((lane >> 4) * 8);
  const unsigned short* arow1 = arow0 + (size_t)16 * IN_DIM;
  // B fragment (32x16 bf16): lane n holds K [bb..bb+7] and [bb+16..bb+23], bb=(lane>=16)?8:0
  const unsigned short* brow0 = W + (size_t)(colBase + (lane & 15)) * IN_DIM + ((lane >> 4) * 8);
  const unsigned short* brow1 = brow0 + (size_t)16 * IN_DIM;

  v8f acc00 = {};   // (M0, N0)
  v8f acc01 = {};   // (M0, N1)
  v8f acc10 = {};   // (M1, N0)
  v8f acc11 = {};   // (M1, N1)
  for (int kb = 0; kb < IN_DIM; kb += 64) {
    // One prefetch per B stream per 64-element body (= one 128B cacheline
    // per row consumed per body), ~1KB lookahead.
    prefetch_wgp(brow0 + kb + 512);
    prefetch_wgp(brow1 + kb + 512);

#pragma unroll
    for (int ks = 0; ks < 64; ks += 32) {
      const int k0 = kb + ks;
      Frag16 a0, a1, b0, b1;
      a0.q[0] = *(const uint4*)(arow0 + k0);
      a0.q[1] = *(const uint4*)(arow0 + k0 + 16);
      a1.q[0] = *(const uint4*)(arow1 + k0);
      a1.q[1] = *(const uint4*)(arow1 + k0 + 16);
      b0.q[0] = *(const uint4*)(brow0 + k0);
      b0.q[1] = *(const uint4*)(brow0 + k0 + 16);
      b1.q[0] = *(const uint4*)(brow1 + k0);
      b1.q[1] = *(const uint4*)(brow1 + k0 + 16);
      acc00 = __builtin_amdgcn_wmma_f32_16x16x32_bf16(false, a0.v, false, b0.v,
                                                      (short)0, acc00, false, false);
      acc01 = __builtin_amdgcn_wmma_f32_16x16x32_bf16(false, a0.v, false, b1.v,
                                                      (short)0, acc01, false, false);
      acc10 = __builtin_amdgcn_wmma_f32_16x16x32_bf16(false, a1.v, false, b0.v,
                                                      (short)0, acc10, false, false);
      acc11 = __builtin_amdgcn_wmma_f32_16x16x32_bf16(false, a1.v, false, b1.v,
                                                      (short)0, acc11, false, false);
    }
  }

  const int n0  = colBase + (lane & 15);
  const int n1  = n0 + 16;
  const float bv0 = bias[n0];
  const float bv1 = bias[n1];
  const int rOff = (lane >> 4) * 8;   // C/D layout: lanes 16-31 hold M = 8..15
#pragma unroll
  for (int i = 0; i < 8; ++i) {
    int r0 = rowBase + rOff + i;        // M-tile 0
    int r1 = r0 + 16;                   // M-tile 1
    if (r0 < M) {
      Y[(size_t)r0 * OUT_DIM + n0] = acc00[i] + bv0;
      Y[(size_t)r0 * OUT_DIM + n1] = acc01[i] + bv1;
    }
    if (r1 < M) {
      Y[(size_t)r1 * OUT_DIM + n0] = acc10[i] + bv0;
      Y[(size_t)r1 * OUT_DIM + n1] = acc11[i] + bv1;
    }
  }
}

// ---------------------------------------------------------------------------
// Row LayerNorm over OUT_DIM, in place. One block (256 threads) per row.
// ---------------------------------------------------------------------------
__global__ __launch_bounds__(256)
void layernorm_kernel(float* __restrict__ Y, const float* __restrict__ g,
                      const float* __restrict__ b) {
  __shared__ float rsum[256];
  __shared__ float rsq[256];
  float* y = Y + (size_t)blockIdx.x * OUT_DIM;
  float s = 0.f, s2 = 0.f;
  for (int d = threadIdx.x; d < OUT_DIM; d += 256) {
    float v = y[d]; s += v; s2 += v * v;
  }
  rsum[threadIdx.x] = s; rsq[threadIdx.x] = s2;
  __syncthreads();
  for (int off = 128; off > 0; off >>= 1) {
    if (threadIdx.x < off) {
      rsum[threadIdx.x] += rsum[threadIdx.x + off];
      rsq[threadIdx.x]  += rsq[threadIdx.x + off];
    }
    __syncthreads();
  }
  float mu  = rsum[0] * (1.0f / OUT_DIM);
  float var = rsq[0] * (1.0f / OUT_DIM) - mu * mu;
  float inv = rsqrtf(var + 1e-5f);
  for (int d = threadIdx.x; d < OUT_DIM; d += 256) {
    y[d] = (y[d] - mu) * inv * g[d] + b[d];
  }
}

// ---------------------------------------------------------------------------
// Attention + prototype distance. One block per (class, query).
// ---------------------------------------------------------------------------
__global__ __launch_bounds__(256)
void attention_kernel(const float* __restrict__ q_k, const float* __restrict__ q_v,
                      const float* __restrict__ s_k, const float* __restrict__ s_v,
                      float* __restrict__ out) {
  const int c = blockIdx.x;   // class
  const int q = blockIdx.y;   // query
  __shared__ float attn[SEQ_LEN][KM];
  __shared__ float red[256];

  const float* qk = q_k + (size_t)q * SEQ_LEN * OUT_DIM;
  const float* ck = s_k + (size_t)c * KM * OUT_DIM;   // rows c*40 .. c*40+39
  const float inv_sqrt_d = 0.029462783f;              // 1/sqrt(1152)

  // scores[l][m] = <qk[l], ck[m]> / sqrt(D)
  for (int idx = threadIdx.x; idx < SEQ_LEN * KM; idx += blockDim.x) {
    int l = idx / KM, m = idx - l * KM;
    const float4* a  = (const float4*)(qk + (size_t)l * OUT_DIM);
    const float4* bb = (const float4*)(ck + (size_t)m * OUT_DIM);
    float acc = 0.f;
    for (int k4 = 0; k4 < OUT_DIM / 4; ++k4) {
      float4 av = a[k4], bv = bb[k4];
      acc = fmaf(av.x, bv.x, acc); acc = fmaf(av.y, bv.y, acc);
      acc = fmaf(av.z, bv.z, acc); acc = fmaf(av.w, bv.w, acc);
    }
    attn[l][m] = acc * inv_sqrt_d;
  }
  __syncthreads();

  // softmax across all 40 support frames per query frame
  if (threadIdx.x < SEQ_LEN) {
    int l = threadIdx.x;
    float mx = attn[l][0];
    for (int m = 1; m < KM; ++m) mx = fmaxf(mx, attn[l][m]);
    float s = 0.f;
    for (int m = 0; m < KM; ++m) { float e = __expf(attn[l][m] - mx); attn[l][m] = e; s += e; }
    float inv = 1.0f / s;
    for (int m = 0; m < KM; ++m) attn[l][m] *= inv;
  }
  __syncthreads();

  // distance = sum_{l,d} (qv[l,d] - sum_m attn[l][m] * cv[m,d])^2
  const float* qv = q_v + (size_t)q * SEQ_LEN * OUT_DIM;
  const float* cv = s_v + (size_t)c * KM * OUT_DIM;
  float part = 0.f;
  for (int idx = threadIdx.x; idx < SEQ_LEN * OUT_DIM; idx += blockDim.x) {
    int l = idx / OUT_DIM, d = idx - l * OUT_DIM;
    float p = 0.f;
#pragma unroll 8
    for (int m = 0; m < KM; ++m)
      p = fmaf(attn[l][m], cv[(size_t)m * OUT_DIM + d], p);
    float diff = qv[idx] - p;
    part = fmaf(diff, diff, part);
  }
  red[threadIdx.x] = part;
  __syncthreads();
  for (int off = 128; off > 0; off >>= 1) {
    if (threadIdx.x < off) red[threadIdx.x] += red[threadIdx.x + off];
    __syncthreads();
  }
  if (threadIdx.x == 0)
    out[q * WAY + c] = -red[0] * (1.0f / 96.0f);   // 96 = sqrt(1152 * 8)
}

// ---------------------------------------------------------------------------
extern "C" void kernel_launch(void* const* d_in, const int* in_sizes, int n_in,
                              void* d_out, int out_size, void* d_ws, size_t ws_size,
                              hipStream_t stream) {
  (void)in_sizes; (void)n_in; (void)out_size; (void)ws_size;
  const float* support = (const float*)d_in[0];
  // d_in[1] = support_labels: sorted class-major, already the reshape order -> unused
  const float* queries = (const float*)d_in[2];
  const float* sk_W = (const float*)d_in[3];
  const float* sk_b = (const float*)d_in[4];
  const float* sv_W = (const float*)d_in[5];
  const float* sv_b = (const float*)d_in[6];
  const float* qk_W = (const float*)d_in[7];
  const float* qk_b = (const float*)d_in[8];
  const float* qv_W = (const float*)d_in[9];
  const float* qv_b = (const float*)d_in[10];
  const float* ln_g = (const float*)d_in[11];
  const float* ln_b = (const float*)d_in[12];
  const float* pe   = (const float*)d_in[13];
  float* out = (float*)d_out;

  // Workspace partition (~31 MB)
  char* ws = (char*)d_ws;
  size_t off = 0;
  auto alloc = [&](size_t bytes) -> void* {
    void* p = ws + off;
    off = (off + bytes + 255) & ~(size_t)255;
    return p;
  };
  unsigned short* Xs  = (unsigned short*)alloc((size_t)S_ROWS_PAD * IN_DIM * 2);
  unsigned short* Xq  = (unsigned short*)alloc((size_t)Q_ROWS * IN_DIM * 2);
  unsigned short* Wsk = (unsigned short*)alloc((size_t)OUT_DIM * IN_DIM * 2);
  unsigned short* Wsv = (unsigned short*)alloc((size_t)OUT_DIM * IN_DIM * 2);
  unsigned short* Wqk = (unsigned short*)alloc((size_t)OUT_DIM * IN_DIM * 2);
  unsigned short* Wqv = (unsigned short*)alloc((size_t)OUT_DIM * IN_DIM * 2);
  float* SK = (float*)alloc((size_t)S_ROWS_PAD * OUT_DIM * 4);
  float* SV = (float*)alloc((size_t)S_ROWS_PAD * OUT_DIM * 4);
  float* QK = (float*)alloc((size_t)Q_ROWS * OUT_DIM * 4);
  float* QV = (float*)alloc((size_t)Q_ROWS * OUT_DIM * 4);

  // 1) activations -> bf16 (+PE, zero-padded rows)
  {
    int ns = S_ROWS_PAD * IN_DIM;
    prep_x_kernel<<<(ns + 255) / 256, 256, 0, stream>>>(support, pe, Xs, S_ROWS, S_ROWS_PAD);
    int nq = Q_ROWS * IN_DIM;
    prep_x_kernel<<<(nq + 255) / 256, 256, 0, stream>>>(queries, pe, Xq, Q_ROWS, Q_ROWS);
  }
  // 2) weights -> bf16
  {
    int wn = OUT_DIM * IN_DIM;
    int gb = (wn + 255) / 256;
    cvt_bf16_kernel<<<gb, 256, 0, stream>>>(sk_W, Wsk, wn);
    cvt_bf16_kernel<<<gb, 256, 0, stream>>>(sv_W, Wsv, wn);
    cvt_bf16_kernel<<<gb, 256, 0, stream>>>(qk_W, Wqk, wn);
    cvt_bf16_kernel<<<gb, 256, 0, stream>>>(qv_W, Wqv, wn);
  }
  // 3) WMMA projections
  {
    dim3 gs(OUT_DIM / 128, S_ROWS_PAD / 32);   // (9, 7)
    dim3 gq(OUT_DIM / 128, Q_ROWS / 32);       // (9, 25)
    gemm_bf16_kernel<<<gs, 128, 0, stream>>>(Xs, Wsk, sk_b, SK, S_ROWS);
    gemm_bf16_kernel<<<gs, 128, 0, stream>>>(Xs, Wsv, sv_b, SV, S_ROWS);
    gemm_bf16_kernel<<<gq, 128, 0, stream>>>(Xq, Wqk, qk_b, QK, Q_ROWS);
    gemm_bf16_kernel<<<gq, 128, 0, stream>>>(Xq, Wqv, qv_b, QV, Q_ROWS);
  }
  // 4) LayerNorm the key projections
  layernorm_kernel<<<S_ROWS, 256, 0, stream>>>(SK, ln_g, ln_b);
  layernorm_kernel<<<Q_ROWS, 256, 0, stream>>>(QK, ln_g, ln_b);
  // 5) attention + distances
  {
    dim3 ga(WAY, NQ);
    attention_kernel<<<ga, 256, 0, stream>>>(QK, QV, SK, SV, out);
  }
}
